// PhpNetASTGraphDependenceTokens_62010737820207
// MI455X (gfx1250) — compile-verified, matrix operand order checked by reading, not run on hardware
//
#include <hip/hip_runtime.h>
#include <hip/hip_bf16.h>

typedef __attribute__((ext_vector_type(16))) __bf16 v16bf;
typedef __attribute__((ext_vector_type(8)))  __bf16 v8bf;
typedef __attribute__((ext_vector_type(8)))  float  v8f;

__device__ __forceinline__ v16bf ld_frag(const __bf16* row, int kb)
{
    // lane's fragment = two contiguous 16B chunks: K = kb+{0..7}, kb+16+{0..7}
    const v8bf lo = *(const v8bf*)(row + kb);
    const v8bf hi = *(const v8bf*)(row + kb + 16);
    return __builtin_shufflevector(lo, hi,
        0,1,2,3,4,5,6,7,8,9,10,11,12,13,14,15);
}

// ---------------------------------------------------------------------------
// WMMA GEMM: C[M,N] = A[M,Kp](bf16, row pitch lda=Kp) @ B[Kp,N](bf16) + bias
// Kp multiple of 32; K-padding in A (cols) and B (rows) is ZERO -> no remainder.
// M/N edge handling by ADDRESS CLAMPING (garbage lands only in discarded C
// rows/cols), so staging is completely branch-free.
// Block = 256 threads (8 waves), 128x64 tile, 32x32 per wave = 4 WMMA.
// Double-buffered LDS: one __syncthreads per K-step; next tile's global loads
// overlap current tile's WMMAs.
// ---------------------------------------------------------------------------
__global__ __launch_bounds__(256) void wmma_gemm_kernel(
    const __bf16* __restrict__ A, int lda,
    const __bf16* __restrict__ Bm,
    const float* __restrict__ bias,
    float* __restrict__ Cf, __bf16* __restrict__ Cb, int ldcb,
    int M, int Nc, int Kp, int relu)
{
    __shared__ alignas(16) __bf16 As [2][128][40];  // A tile rows    (80B pitch)
    __shared__ alignas(16) __bf16 BsT[2][ 64][40];  // B tile COLUMNS (80B pitch)

    const int tid  = threadIdx.x;
    const int wave = tid >> 5;
    const int lane = tid & 31;
    const int wm   = wave & 3;       // 4 row-groups of 32
    const int wn   = wave >> 2;      // 2 col-groups of 32
    const int half = lane >> 4;
    const int lr   = lane & 15;

    const int mBase = blockIdx.x * 128;
    const int nBase = blockIdx.y * 64;

    v8f acc00 = {}, acc01 = {}, acc10 = {}, acc11 = {};

    // staging assignments
    const int aRow = tid >> 1;            // 128 rows, 2 threads/row, 16 elems
    const int aCol = (tid & 1) * 16;
    const int bCol = tid & 63;            // 64 columns, 4 k-groups of 8
    const int bKg  = (tid >> 6) * 8;

    // clamped, loop-carried global pointers (rows 64B aligned: lda % 32 == 0)
    const int aRowC = min(mBase + aRow, M - 1);
    const int bColC = min(nBase + bCol, Nc - 1);
    const __bf16* aIter = A + (size_t)aRowC * lda + aCol;
    const __bf16* bIter = Bm + (size_t)bKg * Nc + bColC;

    const int kb = half * 8;

    v8bf ar0, ar1;
    union { __bf16 b[8]; v8bf v; } bu;

    auto load_regs = [&]() {
        ar0 = *(const v8bf*)(aIter);
        ar1 = *(const v8bf*)(aIter + 8);
        #pragma unroll
        for (int j = 0; j < 8; ++j) bu.b[j] = bIter[(size_t)j * Nc];
        aIter += 32;
        bIter += (size_t)32 * Nc;
    };
    auto store_lds = [&](int buf) {
        *(v8bf*)&As[buf][aRow][aCol]     = ar0;
        *(v8bf*)&As[buf][aRow][aCol + 8] = ar1;
        *(v8bf*)&BsT[buf][bCol][bKg]     = bu.v;
    };
    auto compute = [&](int buf) {
        const v16bf a0 = ld_frag(&As[buf][wm * 32 + lr][0],      kb);
        const v16bf a1 = ld_frag(&As[buf][wm * 32 + 16 + lr][0], kb);
        const v16bf b0 = ld_frag(&BsT[buf][wn * 32 + lr][0],      kb);
        const v16bf b1 = ld_frag(&BsT[buf][wn * 32 + 16 + lr][0], kb);
        acc00 = __builtin_amdgcn_wmma_f32_16x16x32_bf16(false, a0, false, b0,
                                                        (short)0, acc00, false, false);
        acc01 = __builtin_amdgcn_wmma_f32_16x16x32_bf16(false, a0, false, b1,
                                                        (short)0, acc01, false, false);
        acc10 = __builtin_amdgcn_wmma_f32_16x16x32_bf16(false, a1, false, b0,
                                                        (short)0, acc10, false, false);
        acc11 = __builtin_amdgcn_wmma_f32_16x16x32_bf16(false, a1, false, b1,
                                                        (short)0, acc11, false, false);
    };

    const int T = Kp >> 5;
    int buf = 0;
    load_regs();
    for (int it = 0; it < T - 1; ++it) {
        store_lds(buf);
        load_regs();           // next tile's global loads overlap compute below
        __syncthreads();
        compute(buf);
        buf ^= 1;
    }
    store_lds(buf);
    __syncthreads();
    compute(buf);

    // ---- store: C/D layout: VGPR r -> row r + 8*half, col = lane%16 ----
    const int row0 = mBase + wm * 32 + half * 8;
    const int c0   = nBase + wn * 32 + lr;
    const int c1   = c0 + 16;
    const float bi0 = (bias && c0 < Nc) ? bias[c0] : 0.0f;
    const float bi1 = (bias && c1 < Nc) ? bias[c1] : 0.0f;

    auto st = [&](int row, int col, float v, float bv) {
        if (row >= M || col >= Nc) return;
        v += bv;
        if (relu) v = fmaxf(v, 0.0f);
        if (Cf) Cf[(size_t)row * Nc + col] = v;
        if (Cb) Cb[(size_t)row * ldcb + col] = (__bf16)v;
    };

    #pragma unroll
    for (int r = 0; r < 8; ++r) {
        st(row0 + r,      c0, acc00[r], bi0);
        st(row0 + r,      c1, acc01[r], bi1);
        st(row0 + 16 + r, c0, acc10[r], bi0);
        st(row0 + 16 + r, c1, acc11[r], bi1);
    }
}

// ---------------------------------------------------------------------------
// Embedding gather, 4 features per thread; hb has padded pitch ldhb.
// ---------------------------------------------------------------------------
__global__ void embed_kernel(const int* __restrict__ x, const float* __restrict__ emb,
                             float* __restrict__ h, __bf16* __restrict__ hb,
                             long total4, int T, int D, int ldhb)
{
    long i = (long)blockIdx.x * blockDim.x + threadIdx.x;
    if (i >= total4) return;
    const int tdq = T * (D / 4);
    const long n = i / tdq;
    const int  r = (int)(i - n * tdq);
    const int  t = r / (D / 4);
    const int  d4 = (r - t * (D / 4)) * 4;
    const int tok = x[n * T + t];
    const float4 v = *(const float4*)(emb + (size_t)tok * D + d4);
    ((float4*)h)[i] = v;
    union { __bf16 b[4]; uint2 u; } p;
    p.b[0] = (__bf16)v.x; p.b[1] = (__bf16)v.y;
    p.b[2] = (__bf16)v.z; p.b[3] = (__bf16)v.w;
    *(uint2*)(hb + n * ldhb + (size_t)r * 4) = p.u;
}

// float -> bf16, flat, 4 elements per thread
__global__ void f2b_kernel(const float* __restrict__ x, __bf16* __restrict__ y, long n4)
{
    long i = (long)blockIdx.x * blockDim.x + threadIdx.x;
    if (i >= n4) return;
    const float4 v = ((const float4*)x)[i];
    union { __bf16 b[4]; uint2 u; } p;
    p.b[0] = (__bf16)v.x; p.b[1] = (__bf16)v.y;
    p.b[2] = (__bf16)v.z; p.b[3] = (__bf16)v.w;
    ((uint2*)y)[i] = p.u;
}

// float[rows,w] -> bf16[rows,ldd] (pitch-changing)
__global__ void f2b_pitch_kernel(const float* __restrict__ x, __bf16* __restrict__ y,
                                 long total4, int wq, int ldd)
{
    long i = (long)blockIdx.x * blockDim.x + threadIdx.x;
    if (i >= total4) return;
    const long row = i / wq;
    const int  c4  = (int)(i - row * wq) * 4;
    const float4 v = *(const float4*)(x + row * (size_t)(wq * 4) + c4);
    union { __bf16 b[4]; uint2 u; } p;
    p.b[0] = (__bf16)v.x; p.b[1] = (__bf16)v.y;
    p.b[2] = (__bf16)v.z; p.b[3] = (__bf16)v.w;
    *(uint2*)(y + row * (size_t)ldd + c4) = p.u;
}

// ---------------------------------------------------------------------------
// Edge scatter-add: agg[dst] += m[src]; one uint2 (4xbf16) load + 4 f32 atomics
// ---------------------------------------------------------------------------
__global__ void scatter_add_kernel(const __bf16* __restrict__ mb, const int* __restrict__ ei,
                                   float* __restrict__ agg, int E, int H)
{
    const long total = (long)E * (H / 4);
    long i = (long)blockIdx.x * blockDim.x + threadIdx.x;
    if (i >= total) return;
    const int perE = H / 4;
    const long e  = i / perE;
    const int  f4 = (int)(i - e * perE) * 4;
    const int src = ei[e];
    const int dst = ei[(long)E + e];
    union { uint2 u; __bf16 b[4]; } p;
    p.u = *(const uint2*)(mb + (size_t)src * H + f4);
    float* ap = agg + (size_t)dst * H + f4;
    atomicAdd(ap + 0, (float)p.b[0]);
    atomicAdd(ap + 1, (float)p.b[1]);
    atomicAdd(ap + 2, (float)p.b[2]);
    atomicAdd(ap + 3, (float)p.b[3]);
}

// ---------------------------------------------------------------------------
// GRU gate pointwise, 4 features per thread; hb has padded pitch ldhb.
// ---------------------------------------------------------------------------
__device__ __forceinline__ float sigmoidf_(float x) {
    return 1.0f / (1.0f + __expf(-x));
}

__global__ void gru_kernel(const float* __restrict__ gi, const float* __restrict__ gh,
                           float* __restrict__ h, __bf16* __restrict__ hb,
                           long total4, int H, int ldhb)
{
    long i = (long)blockIdx.x * blockDim.x + threadIdx.x;
    if (i >= total4) return;
    const int hq = H / 4;
    const long n = i / hq;
    const int  j4 = (int)(i - n * hq) * 4;
    const long base = n * (3L * H) + j4;

    union F4 { float4 v; float f[4]; };
    F4 ir, iz, in_, hr, hz, hn, hv, ho;
    ir.v  = *(const float4*)(gi + base);
    iz.v  = *(const float4*)(gi + base + H);
    in_.v = *(const float4*)(gi + base + 2 * H);
    hr.v  = *(const float4*)(gh + base);
    hz.v  = *(const float4*)(gh + base + H);
    hn.v  = *(const float4*)(gh + base + 2 * H);
    hv.v  = ((const float4*)h)[i];

    union { __bf16 b[4]; uint2 u; } p;
    #pragma unroll
    for (int q = 0; q < 4; ++q) {
        const float r = sigmoidf_(ir.f[q] + hr.f[q]);
        const float z = sigmoidf_(iz.f[q] + hz.f[q]);
        const float nn = tanhf(in_.f[q] + r * hn.f[q]);
        const float o = (1.0f - z) * nn + z * hv.f[q];
        ho.f[q] = o;
        p.b[q] = (__bf16)o;
    }
    ((float4*)h)[i] = ho.v;
    *(uint2*)(hb + n * (size_t)ldhb + j4) = p.u;
}

// ---------------------------------------------------------------------------
// Segment-max pooling with relu (int atomicMax on nonneg float bits).
// ---------------------------------------------------------------------------
__global__ void pool_kernel(const float* __restrict__ h, const int* __restrict__ bidx,
                            float* __restrict__ pool, long total4, int H,
                            int colOff, int poolStride)
{
    long i = (long)blockIdx.x * blockDim.x + threadIdx.x;
    if (i >= total4) return;
    const int hq = H / 4;
    const long n = i / hq;
    const int  j4 = (int)(i - n * hq) * 4;
    union { float4 v; float f[4]; } u;
    u.v = ((const float4*)h)[i];
    const int b = bidx[n];
    int* pp = (int*)&pool[(size_t)b * poolStride + colOff + j4];
    #pragma unroll
    for (int q = 0; q < 4; ++q)
        atomicMax(pp + q, __float_as_int(fmaxf(u.f[q], 0.0f)));
}

// ---------------------------------------------------------------------------
// Host-side orchestration
// ---------------------------------------------------------------------------
namespace {

constexpr long kN = 100000;
constexpr long kE = 400000;
constexpr int  kT = 20;
constexpr int  kB = 128;
constexpr int  kH = 400;
constexpr int  kD = 20;
constexpr int  kHp  = 416;   // roundup32(400)
constexpr int  kC3p = 1216;  // roundup32(1200)

inline void gemm(const __bf16* A, int lda, const __bf16* B, const float* bias,
                 float* Cf, __bf16* Cb, int ldcb, int M, int N, int Kp, int relu,
                 hipStream_t s)
{
    dim3 grid((M + 127) / 128, (N + 63) / 64);
    wmma_gemm_kernel<<<grid, 256, 0, s>>>(A, lda, B, bias, Cf, Cb, ldcb,
                                          M, N, Kp, relu);
}

inline void f2b(const float* x, __bf16* y, long n, hipStream_t s)
{
    const long n4 = n / 4;
    f2b_kernel<<<(unsigned)((n4 + 255) / 256), 256, 0, s>>>(x, y, n4);
}

inline void f2b_pitch(const float* x, __bf16* y, long rows, int w, int ldd,
                      hipStream_t s)
{
    const long t4 = rows * (w / 4);
    f2b_pitch_kernel<<<(unsigned)((t4 + 255) / 256), 256, 0, s>>>(x, y, t4, w / 4, ldd);
}

} // namespace

extern "C" void kernel_launch(void* const* d_in, const int* in_sizes, int n_in,
                              void* d_out, int out_size, void* d_ws, size_t ws_size,
                              hipStream_t stream)
{
    (void)in_sizes; (void)n_in; (void)out_size; (void)ws_size;

    char* ws = (char*)d_ws;
    size_t off = 0;
    auto take = [&](size_t bytes) -> void* {
        void* p = ws + off;
        off = (off + bytes + 255) & ~(size_t)255;
        return p;
    };

    // ---- zero-initialized region (K-padding must be zero) ----
    __bf16* hb    = (__bf16*)take(kN * kHp * 2);        // [N,416]
    __bf16* aggb  = (__bf16*)take(kN * kHp * 2);        // [N,416]
    __bf16* poolb = (__bf16*)take((long)kB * kC3p * 2); // [128,1216]
    __bf16* wb    = (__bf16*)take(3L * kHp * kH * 2);   // 3 x [416,400]
    __bf16* wihb  = (__bf16*)take((long)kHp * 3 * kH * 2); // [416,1200]
    __bf16* whhb  = (__bf16*)take((long)kHp * 3 * kH * 2); // [416,1200]
    __bf16* wl1b  = (__bf16*)take(1216L * 1000 * 2);
    __bf16* wl2b  = (__bf16*)take(1024L * 750 * 2);
    __bf16* wl21b = (__bf16*)take( 768L * 500 * 2);
    __bf16* wl22b = (__bf16*)take( 512L * 400 * 2);
    __bf16* wl3b  = (__bf16*)take( 416L * 100 * 2);
    __bf16* wl4b  = (__bf16*)take( 128L * 4   * 2);
    __bf16* m1b   = (__bf16*)take((long)kB * 1024 * 2);
    __bf16* m2b   = (__bf16*)take((long)kB * 768  * 2);
    __bf16* m3b   = (__bf16*)take((long)kB * 512  * 2);
    __bf16* m4b   = (__bf16*)take((long)kB * 416  * 2);
    __bf16* m5b   = (__bf16*)take((long)kB * 128  * 2);
    const size_t zlen = off;

    // ---- no-init region ----
    float*  h    = (float*) take(kN * kH * 4);
    __bf16* mb   = (__bf16*)take(kN * kH * 2);
    float*  agg  = (float*) take(kN * kH * 4);
    float*  gi   = (float*) take(kN * 3L * kH * 4);
    float*  gh   = (float*) take(kN * 3L * kH * 4);
    float*  pool = (float*) take((long)kB * 3 * kH * 4);

    hipMemsetAsync(ws, 0, zlen, stream);                        // zero all padding
    hipMemsetAsync(pool, 0, (size_t)kB * 3 * kH * 4, stream);   // pool identity

    const long nh  = kN * kH;
    const long nh4 = nh / 4;

    // ---- three GNN branches ----
    for (int g = 0; g < 3; ++g) {
        const int*   x   = (const int*)  d_in[g * 9 + 0];
        const int*   ei  = (const int*)  d_in[g * 9 + 1];
        const int*   bix = (const int*)  d_in[g * 9 + 2];
        const float* emb = (const float*)d_in[g * 9 + 3];
        const float* w   = (const float*)d_in[g * 9 + 4];
        const float* wih = (const float*)d_in[g * 9 + 5];
        const float* bih = (const float*)d_in[g * 9 + 6];
        const float* whh = (const float*)d_in[g * 9 + 7];
        const float* bhh = (const float*)d_in[g * 9 + 8];

        for (int l = 0; l < 3; ++l)  // per-layer [400,400] into [416,400]
            f2b(w + (size_t)l * kH * kH, wb + (size_t)l * kHp * kH,
                (long)kH * kH, stream);
        f2b(wih, wihb, (long)kH * 3 * kH, stream);   // rows 0..399 of [416,1200]
        f2b(whh, whhb, (long)kH * 3 * kH, stream);

        const long etot4 = kN * kT * (kD / 4);
        embed_kernel<<<(unsigned)((etot4 + 255) / 256), 256, 0, stream>>>(
            x, emb, h, hb, etot4, kT, kD, kHp);

        for (int layer = 0; layer < 3; ++layer) {
            // m = h @ w[layer]  (bf16 out, pitch 400; feeds edge gather)
            gemm(hb, kHp, wb + (size_t)layer * kHp * kH, nullptr,
                 nullptr, mb, kH, (int)kN, kH, kHp, 0, stream);

            // agg = segment_sum(m[src] -> dst)
            hipMemsetAsync(agg, 0, (size_t)nh * 4, stream);
            const long stot = kE * (kH / 4);
            scatter_add_kernel<<<(unsigned)((stot + 255) / 256), 256, 0, stream>>>(
                mb, ei, agg, (int)kE, kH);
            f2b_pitch(agg, aggb, kN, kH, kHp, stream);

            // gi = agg @ wih + bih ; gh = h @ whh + bhh
            gemm(aggb, kHp, wihb, bih, gi, nullptr, 0, (int)kN, 3 * kH, kHp, 0, stream);
            gemm(hb,   kHp, whhb, bhh, gh, nullptr, 0, (int)kN, 3 * kH, kHp, 0, stream);

            gru_kernel<<<(unsigned)((nh4 + 255) / 256), 256, 0, stream>>>(
                gi, gh, h, hb, nh4, kH, kHp);
        }

        pool_kernel<<<(unsigned)((nh4 + 255) / 256), 256, 0, stream>>>(
            h, bix, pool, nh4, kH, g * kH, 3 * kH);
    }

    // ---- MLP head on concat [128, 1200] ----
    f2b_pitch(pool, poolb, kB, 3 * kH, kC3p, stream);

    const float* wl1  = (const float*)d_in[27]; const float* bl1  = (const float*)d_in[28];
    const float* wl2  = (const float*)d_in[29]; const float* bl2  = (const float*)d_in[30];
    const float* wl21 = (const float*)d_in[31]; const float* bl21 = (const float*)d_in[32];
    const float* wl22 = (const float*)d_in[33]; const float* bl22 = (const float*)d_in[34];
    const float* wl3  = (const float*)d_in[35]; const float* bl3  = (const float*)d_in[36];
    const float* wl4  = (const float*)d_in[37]; const float* bl4  = (const float*)d_in[38];

    f2b(wl1,  wl1b,  1200L * 1000, stream);
    gemm(poolb, kC3p, wl1b, bl1, nullptr, m1b, 1024, kB, 1000, kC3p, 1, stream);

    f2b(wl2,  wl2b,  1000L * 750, stream);
    gemm(m1b, 1024, wl2b, bl2, nullptr, m2b, 768, kB, 750, 1024, 1, stream);

    f2b(wl21, wl21b, 750L * 500, stream);
    gemm(m2b, 768, wl21b, bl21, nullptr, m3b, 512, kB, 500, 768, 1, stream);

    f2b(wl22, wl22b, 500L * 400, stream);
    gemm(m3b, 512, wl22b, bl22, nullptr, m4b, 416, kB, 400, 512, 1, stream);

    f2b(wl3,  wl3b,  400L * 100, stream);
    gemm(m4b, 416, wl3b, bl3, nullptr, m5b, 128, kB, 100, 416, 1, stream);

    f2b(wl4,  wl4b,  100L * 4, stream);
    gemm(m5b, 128, wl4b, bl4, (float*)d_out, nullptr, 0, kB, 4, 128, 1, stream);
}
